// SqueezeformerBlock_83107617177636
// MI455X (gfx1250) — compile-verified, matrix operand order checked by reading, not run on hardware
//
#include <hip/hip_runtime.h>
#include <hip/hip_bf16.h>
#include <math.h>

typedef __attribute__((ext_vector_type(16))) _Float16 v16h;
typedef __attribute__((ext_vector_type(8)))  _Float16 v8h;
typedef __attribute__((ext_vector_type(8)))  float    v8f;

// 16-byte int vector matching the async-LDS builtin's parameter type,
// qualified with the AMDGPU global (AS1) / LDS (AS3) address spaces.
typedef int v4i_t __attribute__((vector_size(16)));
typedef __attribute__((address_space(1))) v4i_t g_v4i;
typedef __attribute__((address_space(3))) v4i_t l_v4i;

namespace sqz {

constexpr int Bc = 16, Tc = 512, Dc = 512, Hc = 8, DIc = 2048, KC = 31, Gc = 8;
constexpr int DHc = Dc / Hc;       // 64
constexpr int Mc = Bc * Tc;        // 8192
constexpr int LPAD = 1024;         // 2T-1 = 1023, padded
constexpr float EPSc = 1e-5f;

__device__ __forceinline__ float silu_f(float x) { return x / (1.0f + __expf(-x)); }

// Load a 16x32 (MxK) f16 A-fragment (or 32x16 KxN B-fragment from an [N,K]
// row-major buffer). ISA layout: lanes 0-15 hold K={0..7,16..23}, lanes 16-31
// hold K={8..15,24..31} -> two 16-byte contiguous loads per lane.
__device__ __forceinline__ v16h load_frag(const _Float16* __restrict__ base,
                                          int stride, int row0, int k0) {
  int lane = threadIdx.x & 31;
  int r    = lane & 15;
  int sel  = (lane & 16) ? 8 : 0;
  const _Float16* p = base + (size_t)(row0 + r) * stride + (k0 + sel);
  v8h lo = *(const v8h*)p;
  v8h hi = *(const v8h*)(p + 16);
  v16h out;
#pragma unroll
  for (int i = 0; i < 8; ++i) { out[i] = lo[i]; out[i + 8] = hi[i]; }
  return out;
}

__device__ __forceinline__ v8f wmma16(v16h a, v16h b, v8f c) {
  return __builtin_amdgcn_wmma_f32_16x16x32_f16(false, a, false, b, (short)0, c,
                                                false, false);
}

// ---- CDNA5 async global->LDS staging (ASYNCcnt-tracked DMA) ----------------
__device__ __forceinline__ void async_copy16(const _Float16* gp, _Float16* lp) {
#if __has_builtin(__builtin_amdgcn_global_load_async_to_lds_b128)
  __builtin_amdgcn_global_load_async_to_lds_b128((g_v4i*)gp, (l_v4i*)lp, 0, 0);
#else
  *(v8h*)lp = *(const v8h*)gp;   // synchronous fallback
#endif
}

template <int N>
__device__ __forceinline__ void wait_async() {
#if __has_builtin(__builtin_amdgcn_global_load_async_to_lds_b128)
#if __has_builtin(__builtin_amdgcn_s_wait_asynccnt)
  __builtin_amdgcn_s_wait_asynccnt((short)N);
#else
  asm volatile("s_wait_asynccnt %0" ::"i"(N) : "memory");
#endif
#endif
}

// ---------------------------------------------------------------- GEMM ------
// C[M,N] = A[M,K] (f16, row-major) x Bt[N,K]^T (f16, N-major) [+bias][silu][+res]
// Wave computes a 16x64 strip (4 WMMA tiles sharing one A fragment). The A
// tile is double-buffered through LDS using async DMA; B fragments stream
// from global (weights are L2-resident).
template <int HAS_BIAS, int DO_SILU, int ADD_RES, int OUT_F32, int OUT_F16>
__global__ void __launch_bounds__(256) gemm_wmma(
    const _Float16* __restrict__ A, const _Float16* __restrict__ Bt,
    const float* __restrict__ bias, const float* __restrict__ res,
    float* __restrict__ outF, _Float16* __restrict__ outH,
    int Mm, int Nn, int Kk) {
  __shared__ _Float16 atile[8][2][16 * 32];  // per-wave ping-pong A tiles
  int lane = threadIdx.x & 31;
  int wave = threadIdx.x >> 5;
  int nblk = Nn >> 6;
  long tile = (long)blockIdx.x * 8 + wave;
  long total = (long)(Mm >> 4) * nblk;
  if (tile >= total) return;
  int mt = (int)(tile / nblk);
  int nb = (int)(tile % nblk);
  int m0 = mt << 4;
  int n0 = nb << 6;

  int arow = lane >> 2;          // 0..7
  int asub = (lane & 3) * 8;     // halves 0,8,16,24
  auto issue_a = [&](int k0, int buf) {
    const _Float16* gp0 = A + (size_t)(m0 + arow) * Kk + k0 + asub;
    const _Float16* gp1 = A + (size_t)(m0 + arow + 8) * Kk + k0 + asub;
    async_copy16(gp0, &atile[wave][buf][arow * 32 + asub]);
    async_copy16(gp1, &atile[wave][buf][(arow + 8) * 32 + asub]);
  };

  v8f acc[4];
#pragma unroll
  for (int j = 0; j < 4; ++j) acc[j] = v8f{0, 0, 0, 0, 0, 0, 0, 0};

  issue_a(0, 0);
  int it = 0;
  for (int k0 = 0; k0 < Kk; k0 += 32, ++it) {
    int cur = it & 1;
    if (k0 + 32 < Kk) {
      issue_a(k0 + 32, cur ^ 1);
      __builtin_prefetch(Bt + (size_t)n0 * Kk + k0 + 32, 0, 1);
      wait_async<2>();           // two oldest (current buffer) are complete
    } else {
      wait_async<0>();
    }
    v16h a = load_frag(&atile[wave][cur][0], 32, 0, 0);
#pragma unroll
    for (int j = 0; j < 4; ++j) {
      v16h b = load_frag(Bt, Kk, n0 + 16 * j, k0);
      acc[j] = wmma16(a, b, acc[j]);
    }
  }

  int col = lane & 15;
  int rbase = (lane & 16) ? 8 : 0;
#pragma unroll
  for (int j = 0; j < 4; ++j) {
    int n = n0 + 16 * j + col;
    float bv = HAS_BIAS ? bias[n] : 0.0f;
#pragma unroll
    for (int r = 0; r < 8; ++r) {
      size_t idx = (size_t)(m0 + rbase + r) * Nn + n;
      float v = acc[j][r] + bv;
      if (DO_SILU) v = silu_f(v);
      if (ADD_RES) v += res[idx];
      if (OUT_F32) outF[idx] = v;
      if (OUT_F16) outH[idx] = (_Float16)v;
    }
  }
}

// ------------------------------------------------- attention scores --------
// scores[b,h,t,s] = ((q+u)_t . k_s + (q+v)_t . pe[s-t+T-1]) / sqrt(DH), masked.
// Pos term: 16x31 Toeplitz mini-GEMM per tile, gathered via LDS.
__global__ void __launch_bounds__(256) attn_scores_kernel(
    const _Float16* __restrict__ quh, const _Float16* __restrict__ qvh,
    const _Float16* __restrict__ kh, const _Float16* __restrict__ peh,
    const unsigned char* __restrict__ mask, float* __restrict__ scores) {
  __shared__ float pls[8 * 16 * 32];
  int lane = threadIdx.x & 31;
  int wave = threadIdx.x >> 5;
  long tile = (long)blockIdx.x * 8 + wave;  // ((b*H+h)*32 + tt)*32 + st, exact grid
  int st = (int)(tile & 31);
  int tt = (int)((tile >> 5) & 31);
  int bh = (int)(tile >> 10);
  int h = bh & (Hc - 1);
  int b = bh >> 3;
  int t0 = tt << 4, s0 = st << 4;
  int rowQ = b * Tc + t0;
  int rowK = b * Tc + s0;
  int colh = h * DHc;
  int l0 = s0 - t0 + (Tc - 16);  // pe row base; in [0, 992]

  v8f accC = {0, 0, 0, 0, 0, 0, 0, 0};
  v8f accP0 = {0, 0, 0, 0, 0, 0, 0, 0};
  v8f accP1 = {0, 0, 0, 0, 0, 0, 0, 0};
#pragma unroll
  for (int k0 = 0; k0 < DHc; k0 += 32) {
    v16h au = load_frag(quh, Dc, rowQ, colh + k0);
    v16h bk = load_frag(kh, Dc, rowK, colh + k0);
    accC = wmma16(au, bk, accC);
    v16h av = load_frag(qvh, Dc, rowQ, colh + k0);
    v16h p0 = load_frag(peh, Dc, l0, colh + k0);
    v16h p1 = load_frag(peh, Dc, l0 + 16, colh + k0);
    accP0 = wmma16(av, p0, accP0);
    accP1 = wmma16(av, p1, accP1);
  }

  float* pl = &pls[wave * 512];
  int col = lane & 15;
  int rbase = (lane & 16) ? 8 : 0;
#pragma unroll
  for (int r = 0; r < 8; ++r) {
    pl[(rbase + r) * 32 + col] = accP0[r];
    pl[(rbase + r) * 32 + 16 + col] = accP1[r];
  }
  __syncthreads();
#pragma unroll
  for (int r = 0; r < 8; ++r) {
    int i = rbase + r;
    int c = col - i + 15;  // always in [0,30]
    float v = (accC[r] + pl[i * 32 + c]) * 0.125f;
    if (!mask[b * Tc + s0 + col]) v = -3.4028235e38f;
    scores[((size_t)bh * Tc + (t0 + i)) * Tc + s0 + col] = v;
  }
}

// --------------------------------------------------------- softmax ---------
__global__ void __launch_bounds__(256) softmax_rows(
    const float* __restrict__ scores, _Float16* __restrict__ attn) {
  __shared__ float red[256];
  size_t row = blockIdx.x;
  const float* s = scores + row * Tc;
  int tid = threadIdx.x;
  float a0 = s[tid], a1 = s[tid + 256];
  red[tid] = fmaxf(a0, a1);
  __syncthreads();
  for (int o = 128; o > 0; o >>= 1) {
    if (tid < o) red[tid] = fmaxf(red[tid], red[tid + o]);
    __syncthreads();
  }
  float m = red[0];
  __syncthreads();
  float e0 = __expf(a0 - m), e1 = __expf(a1 - m);
  red[tid] = e0 + e1;
  __syncthreads();
  for (int o = 128; o > 0; o >>= 1) {
    if (tid < o) red[tid] += red[tid + o];
    __syncthreads();
  }
  float inv = 1.0f / red[0];
  attn[row * Tc + tid] = (_Float16)(e0 * inv);
  attn[row * Tc + tid + 256] = (_Float16)(e1 * inv);
}

// --------------------------------------------------------- attn @ V --------
__global__ void __launch_bounds__(256) attn_v_kernel(
    const _Float16* __restrict__ attn, const _Float16* __restrict__ vT,
    _Float16* __restrict__ aoh) {
  int lane = threadIdx.x & 31;
  int wave = threadIdx.x >> 5;
  long tile = (long)blockIdx.x * 8 + wave;  // ((b*H+h)*32 + tt)*4 + nt, exact grid
  int nt = (int)(tile & 3);
  int tt = (int)((tile >> 2) & 31);
  int bh = (int)(tile >> 7);
  int h = bh & (Hc - 1);
  int b = bh >> 3;
  int t0 = tt << 4, n0 = nt << 4;
  v8f acc = {0, 0, 0, 0, 0, 0, 0, 0};
  for (int k0 = 0; k0 < Tc; k0 += 32) {
    v16h a = load_frag(attn, Tc, bh * Tc + t0, k0);
    v16h bb = load_frag(vT, Tc, bh * DHc + n0, k0);
    acc = wmma16(a, bb, acc);
  }
  int col = lane & 15;
  int rbase = (lane & 16) ? 8 : 0;
#pragma unroll
  for (int r = 0; r < 8; ++r) {
    int t = t0 + rbase + r;
    aoh[(size_t)(b * Tc + t) * Dc + h * DHc + n0 + col] = (_Float16)acc[r];
  }
}

// --------------------------------------------------- elementwise helpers ---
__global__ void scale_to_h(const float* __restrict__ in,
                           const float* __restrict__ sc,
                           const float* __restrict__ sh,
                           _Float16* __restrict__ out, int n) {
  int idx = blockIdx.x * blockDim.x + threadIdx.x;
  if (idx >= n) return;
  int d = idx % Dc;
  out[idx] = (_Float16)(sc[d] * in[idx] + sh[d]);
}

__global__ void trans_w16(const float* __restrict__ src,
                          _Float16* __restrict__ dst, int Kd, int Nd) {
  int idx = blockIdx.x * blockDim.x + threadIdx.x;
  if (idx >= Kd * Nd) return;
  int k = idx / Nd, n = idx % Nd;
  dst[(size_t)n * Kd + k] = (_Float16)src[idx];
}

__global__ void relpe_kernel(_Float16* __restrict__ out) {
  int idx = blockIdx.x * blockDim.x + threadIdx.x;
  if (idx >= LPAD * Dc) return;
  int l = idx / Dc, d = idx % Dc;
  float v = 0.0f;
  if (l < 2 * Tc - 1) {
    int i = d >> 1;
    float w = __expf(-9.210340371976184f * (float)(2 * i) / (float)Dc);
    float ang = (float)(l - (Tc - 1)) * w;
    v = (d & 1) ? __cosf(ang) : __sinf(ang);
  }
  out[idx] = (_Float16)v;
}

__global__ void qu_qv_kernel(const _Float16* __restrict__ qh,
                             const float* __restrict__ u,
                             const float* __restrict__ v,
                             _Float16* __restrict__ quh,
                             _Float16* __restrict__ qvh, int n) {
  int idx = blockIdx.x * blockDim.x + threadIdx.x;
  if (idx >= n) return;
  int d = idx % Dc;
  float q = (float)qh[idx];
  quh[idx] = (_Float16)(q + u[d]);
  qvh[idx] = (_Float16)(q + v[d]);
}

__global__ void v_transpose(const _Float16* __restrict__ vh,
                            _Float16* __restrict__ vT, int n) {
  int idx = blockIdx.x * blockDim.x + threadIdx.x;  // [b,h,n,s]
  if (idx >= n) return;
  int s = idx % Tc;
  int nn = (idx / Tc) % DHc;
  int h = (idx / (Tc * DHc)) % Hc;
  int b = idx / (Tc * DHc * Hc);
  vT[idx] = vh[(size_t)(b * Tc + s) * Dc + h * DHc + nn];
}

__global__ void __launch_bounds__(256) layernorm_rows(
    const float* __restrict__ in, const float* __restrict__ g,
    const float* __restrict__ bb, float* __restrict__ out) {
  __shared__ float red[256];
  size_t row = blockIdx.x;
  const float* x = in + row * Dc;
  int tid = threadIdx.x;
  float a0 = x[tid], a1 = x[tid + 256];
  red[tid] = a0 + a1;
  __syncthreads();
  for (int o = 128; o > 0; o >>= 1) {
    if (tid < o) red[tid] += red[tid + o];
    __syncthreads();
  }
  float mu = red[0] * (1.0f / Dc);
  __syncthreads();
  float d0 = a0 - mu, d1 = a1 - mu;
  red[tid] = d0 * d0 + d1 * d1;
  __syncthreads();
  for (int o = 128; o > 0; o >>= 1) {
    if (tid < o) red[tid] += red[tid + o];
    __syncthreads();
  }
  float rs = rsqrtf(red[0] * (1.0f / Dc) + EPSc);
  out[row * Dc + tid] = d0 * rs * g[tid] + bb[tid];
  out[row * Dc + tid + 256] = d1 * rs * g[tid + 256] + bb[tid + 256];
}

__global__ void dwconv_kernel(const _Float16* __restrict__ hin,
                              const float* __restrict__ dw,
                              const float* __restrict__ bdw,
                              const unsigned char* __restrict__ mask,
                              float* __restrict__ out, int n) {
  int idx = blockIdx.x * blockDim.x + threadIdx.x;
  if (idx >= n) return;
  int d = idx % Dc;
  int t = (idx / Dc) % Tc;
  int b = idx / (Dc * Tc);
  float s = 0.0f;
  for (int k = 0; k < KC; ++k) {
    int t2 = t + k - KC / 2;
    if (t2 >= 0 && t2 < Tc && mask[b * Tc + t2])
      s += (float)hin[(size_t)(b * Tc + t2) * Dc + d] * dw[d * KC + k];
  }
  s += bdw[d];
  out[idx] = mask[b * Tc + t] ? s : 0.0f;
}

__global__ void __launch_bounds__(256) groupnorm_silu(
    const float* __restrict__ in, const float* __restrict__ gg,
    const float* __restrict__ gb, _Float16* __restrict__ out) {
  __shared__ float red[256], red2[256];
  int bg = blockIdx.x;
  int g = bg % Gc, b = bg / Gc;
  int tid = threadIdx.x;
  const int CG = Dc / Gc;  // 64
  float s = 0.0f, s2 = 0.0f;
  for (int e = tid; e < CG * Tc; e += 256) {
    int c = e & (CG - 1), t = e >> 6;
    float v = in[(size_t)(b * Tc + t) * Dc + g * CG + c];
    s += v;
    s2 += v * v;
  }
  red[tid] = s;
  red2[tid] = s2;
  __syncthreads();
  for (int o = 128; o > 0; o >>= 1) {
    if (tid < o) { red[tid] += red[tid + o]; red2[tid] += red2[tid + o]; }
    __syncthreads();
  }
  float inv = 1.0f / (float)(CG * Tc);
  float mu = red[0] * inv;
  float var = red2[0] * inv - mu * mu;
  float rs = rsqrtf(var + EPSc);
  for (int e = tid; e < CG * Tc; e += 256) {
    int c = e & (CG - 1), t = e >> 6;
    int d = g * CG + c;
    float v = in[(size_t)(b * Tc + t) * Dc + d];
    float hn = (v - mu) * rs * gg[d] + gb[d];
    out[(size_t)(b * Tc + t) * Dc + d] = (_Float16)silu_f(hn);
  }
}

__global__ void final_out_kernel(const float* __restrict__ x,
                                 const float* __restrict__ y,
                                 const unsigned char* __restrict__ mask,
                                 float* __restrict__ out, int n) {
  int idx = blockIdx.x * blockDim.x + threadIdx.x;
  if (idx >= n) return;
  int bt = idx / Dc;
  float m = mask[bt] ? 1.0f : 0.0f;
  out[idx] = (x[idx] + y[idx]) * m * 0.7071067811865476f;
}

}  // namespace sqz

using namespace sqz;

extern "C" void kernel_launch(void* const* d_in, const int* in_sizes, int n_in,
                              void* d_out, int out_size, void* d_ws,
                              size_t ws_size, hipStream_t stream) {
  (void)in_sizes; (void)n_in; (void)out_size; (void)ws_size;

  // ---- inputs (setup_inputs dict order, nested dicts flattened) ----
  const float* x = (const float*)d_in[0];
  const unsigned char* mask = (const unsigned char*)d_in[1];
  const float* mha_sc = (const float*)d_in[2];
  const float* mha_sh = (const float*)d_in[3];
  const float* Wq = (const float*)d_in[4];
  const float* bq = (const float*)d_in[5];
  const float* Wk = (const float*)d_in[6];
  const float* bk = (const float*)d_in[7];
  const float* Wv = (const float*)d_in[8];
  const float* bv = (const float*)d_in[9];
  const float* Wp = (const float*)d_in[10];
  const float* u_p = (const float*)d_in[11];
  const float* v_p = (const float*)d_in[12];
  const float* Wo = (const float*)d_in[13];
  const float* bo = (const float*)d_in[14];
  const float* ln1g = (const float*)d_in[15];
  const float* ln1b = (const float*)d_in[16];
  const float* f1sc = (const float*)d_in[17];
  const float* f1sh = (const float*)d_in[18];
  const float* W1a = (const float*)d_in[19];
  const float* b1a = (const float*)d_in[20];
  const float* W2a = (const float*)d_in[21];
  const float* b2a = (const float*)d_in[22];
  const float* ln2g = (const float*)d_in[23];
  const float* ln2b = (const float*)d_in[24];
  const float* cvsc = (const float*)d_in[25];
  const float* cvsh = (const float*)d_in[26];
  const float* Wp1 = (const float*)d_in[27];
  const float* bp1 = (const float*)d_in[28];
  const float* dw = (const float*)d_in[29];
  const float* bdw = (const float*)d_in[30];
  const float* gng = (const float*)d_in[31];
  const float* gnb = (const float*)d_in[32];
  const float* Wp2 = (const float*)d_in[33];
  const float* bp2 = (const float*)d_in[34];
  const float* ln3g = (const float*)d_in[35];
  const float* ln3b = (const float*)d_in[36];
  const float* f2sc = (const float*)d_in[37];
  const float* f2sh = (const float*)d_in[38];
  const float* W1b = (const float*)d_in[39];
  const float* b1b = (const float*)d_in[40];
  const float* W2b = (const float*)d_in[41];
  const float* b2b = (const float*)d_in[42];
  const float* ln4g = (const float*)d_in[43];
  const float* ln4b = (const float*)d_in[44];

  // ---- workspace carve-up ----
  char* wsp = (char*)d_ws;
  size_t off = 0;
  auto alloc = [&](size_t bytes) {
    void* p = wsp + off;
    off = (off + bytes + 255) & ~(size_t)255;
    return p;
  };
  const size_t MD = (size_t)Mc * Dc;
  _Float16* x2h  = (_Float16*)alloc(MD * 2);
  _Float16* qh   = (_Float16*)alloc(MD * 2);
  _Float16* quh  = (_Float16*)alloc(MD * 2);
  _Float16* qvh  = (_Float16*)alloc(MD * 2);
  _Float16* kh   = (_Float16*)alloc(MD * 2);
  _Float16* vh   = (_Float16*)alloc(MD * 2);
  _Float16* vTh  = (_Float16*)alloc(MD * 2);
  _Float16* aoh  = (_Float16*)alloc(MD * 2);
  _Float16* cvh  = (_Float16*)alloc(MD * 2);
  _Float16* peih = (_Float16*)alloc((size_t)LPAD * Dc * 2);
  _Float16* peh  = (_Float16*)alloc((size_t)LPAD * Dc * 2);
  _Float16* hbig = (_Float16*)alloc((size_t)Mc * DIc * 2);
  _Float16* attnh = (_Float16*)alloc((size_t)Bc * Hc * Tc * Tc * 2);
  _Float16* WqT = (_Float16*)alloc((size_t)Dc * Dc * 2);
  _Float16* WkT = (_Float16*)alloc((size_t)Dc * Dc * 2);
  _Float16* WvT = (_Float16*)alloc((size_t)Dc * Dc * 2);
  _Float16* WpT = (_Float16*)alloc((size_t)Dc * Dc * 2);
  _Float16* WoT = (_Float16*)alloc((size_t)Dc * Dc * 2);
  _Float16* W1aT = (_Float16*)alloc((size_t)Dc * DIc * 2);
  _Float16* W2aT = (_Float16*)alloc((size_t)Dc * DIc * 2);
  _Float16* Wp1T = (_Float16*)alloc((size_t)Dc * Dc * 2);
  _Float16* Wp2T = (_Float16*)alloc((size_t)Dc * Dc * 2);
  _Float16* W1bT = (_Float16*)alloc((size_t)Dc * DIc * 2);
  _Float16* W2bT = (_Float16*)alloc((size_t)Dc * DIc * 2);
  float* scores = (float*)alloc((size_t)Bc * Hc * Tc * Tc * 4);
  float* bufA = (float*)alloc(MD * 4);
  float* bufB = (float*)alloc(MD * 4);
  float* bufC = (float*)alloc(MD * 4);

  auto ew = [](size_t n) { return dim3((unsigned)((n + 255) / 256)); };
  auto gemm_grid = [](int Mm, int Nn) {
    long tiles = (long)(Mm / 16) * (Nn / 64);
    return dim3((unsigned)((tiles + 7) / 8));
  };
  const dim3 blk(256);

  // ---- prep: weight transposes + relative PE table ----
  trans_w16<<<ew((size_t)Dc * Dc), blk, 0, stream>>>(Wq, WqT, Dc, Dc);
  trans_w16<<<ew((size_t)Dc * Dc), blk, 0, stream>>>(Wk, WkT, Dc, Dc);
  trans_w16<<<ew((size_t)Dc * Dc), blk, 0, stream>>>(Wv, WvT, Dc, Dc);
  trans_w16<<<ew((size_t)Dc * Dc), blk, 0, stream>>>(Wp, WpT, Dc, Dc);
  trans_w16<<<ew((size_t)Dc * Dc), blk, 0, stream>>>(Wo, WoT, Dc, Dc);
  trans_w16<<<ew((size_t)Dc * DIc), blk, 0, stream>>>(W1a, W1aT, Dc, DIc);
  trans_w16<<<ew((size_t)Dc * DIc), blk, 0, stream>>>(W2a, W2aT, DIc, Dc);
  trans_w16<<<ew((size_t)Dc * Dc), blk, 0, stream>>>(Wp1, Wp1T, Dc, Dc);
  trans_w16<<<ew((size_t)Dc * Dc), blk, 0, stream>>>(Wp2, Wp2T, Dc, Dc);
  trans_w16<<<ew((size_t)Dc * DIc), blk, 0, stream>>>(W1b, W1bT, Dc, DIc);
  trans_w16<<<ew((size_t)Dc * DIc), blk, 0, stream>>>(W2b, W2bT, DIc, Dc);
  relpe_kernel<<<ew((size_t)LPAD * Dc), blk, 0, stream>>>(peih);

  // ---- MHA ----
  scale_to_h<<<ew(MD), blk, 0, stream>>>(x, mha_sc, mha_sh, x2h, (int)MD);
  gemm_wmma<1, 0, 0, 0, 1><<<gemm_grid(Mc, Dc), blk, 0, stream>>>(
      x2h, WqT, bq, nullptr, nullptr, qh, Mc, Dc, Dc);
  gemm_wmma<1, 0, 0, 0, 1><<<gemm_grid(Mc, Dc), blk, 0, stream>>>(
      x2h, WkT, bk, nullptr, nullptr, kh, Mc, Dc, Dc);
  gemm_wmma<1, 0, 0, 0, 1><<<gemm_grid(Mc, Dc), blk, 0, stream>>>(
      x2h, WvT, bv, nullptr, nullptr, vh, Mc, Dc, Dc);
  qu_qv_kernel<<<ew(MD), blk, 0, stream>>>(qh, u_p, v_p, quh, qvh, (int)MD);
  v_transpose<<<ew(MD), blk, 0, stream>>>(vh, vTh, (int)MD);
  gemm_wmma<0, 0, 0, 0, 1><<<gemm_grid(LPAD, Dc), blk, 0, stream>>>(
      peih, WpT, nullptr, nullptr, nullptr, peh, LPAD, Dc, Dc);
  attn_scores_kernel<<<dim3(Bc * Hc * 32 * 32 / 8), blk, 0, stream>>>(
      quh, qvh, kh, peh, mask, scores);
  softmax_rows<<<dim3(Bc * Hc * Tc), blk, 0, stream>>>(scores, attnh);
  attn_v_kernel<<<dim3(Bc * Hc * 32 * 4 / 8), blk, 0, stream>>>(attnh, vTh, aoh);
  gemm_wmma<1, 0, 1, 1, 0><<<gemm_grid(Mc, Dc), blk, 0, stream>>>(
      aoh, WoT, bo, x, bufA, nullptr, Mc, Dc, Dc);
  layernorm_rows<<<dim3(Mc), blk, 0, stream>>>(bufA, ln1g, ln1b, bufB);

  // ---- FFN1 ----
  scale_to_h<<<ew(MD), blk, 0, stream>>>(bufB, f1sc, f1sh, x2h, (int)MD);
  gemm_wmma<1, 1, 0, 0, 1><<<gemm_grid(Mc, DIc), blk, 0, stream>>>(
      x2h, W1aT, b1a, nullptr, nullptr, hbig, Mc, DIc, Dc);
  gemm_wmma<1, 0, 1, 1, 0><<<gemm_grid(Mc, Dc), blk, 0, stream>>>(
      hbig, W2aT, b2a, bufB, bufA, nullptr, Mc, Dc, DIc);
  layernorm_rows<<<dim3(Mc), blk, 0, stream>>>(bufA, ln2g, ln2b, bufB);

  // ---- Conv module ----
  scale_to_h<<<ew(MD), blk, 0, stream>>>(bufB, cvsc, cvsh, x2h, (int)MD);
  gemm_wmma<1, 1, 0, 0, 1><<<gemm_grid(Mc, Dc), blk, 0, stream>>>(
      x2h, Wp1T, bp1, nullptr, nullptr, hbig, Mc, Dc, Dc);
  dwconv_kernel<<<ew(MD), blk, 0, stream>>>(hbig, dw, bdw, mask, bufC, (int)MD);
  groupnorm_silu<<<dim3(Bc * Gc), blk, 0, stream>>>(bufC, gng, gnb, cvh);
  gemm_wmma<1, 0, 1, 1, 0><<<gemm_grid(Mc, Dc), blk, 0, stream>>>(
      cvh, Wp2T, bp2, bufB, bufA, nullptr, Mc, Dc, Dc);
  layernorm_rows<<<dim3(Mc), blk, 0, stream>>>(bufA, ln3g, ln3b, bufB);

  // ---- FFN2 ----
  scale_to_h<<<ew(MD), blk, 0, stream>>>(bufB, f2sc, f2sh, x2h, (int)MD);
  gemm_wmma<1, 1, 0, 0, 1><<<gemm_grid(Mc, DIc), blk, 0, stream>>>(
      x2h, W1bT, b1b, nullptr, nullptr, hbig, Mc, DIc, Dc);
  gemm_wmma<1, 0, 1, 1, 0><<<gemm_grid(Mc, Dc), blk, 0, stream>>>(
      hbig, W2bT, b2b, bufB, bufA, nullptr, Mc, Dc, DIc);
  layernorm_rows<<<dim3(Mc), blk, 0, stream>>>(bufA, ln4g, ln4b, bufB);

  // ---- final residual + mask + 1/sqrt(2) ----
  final_out_kernel<<<ew(MD), blk, 0, stream>>>(x, bufB, mask, (float*)d_out,
                                               (int)MD);
}
